// TransformerEncoder_35536559407329
// MI455X (gfx1250) — compile-verified, matrix operand order checked by reading, not run on hardware
//
#include <hip/hip_runtime.h>
#include <hip/hip_bf16.h>

// ---------------- problem constants ----------------
#define DD      512            // model dim
#define BATCH   4
#define SEQ     1024
#define NTOK    (BATCH*SEQ)    // 4096 tokens
#define NHEAD   8
#define DHEAD   64
#define NEXP    32
#define RRANK   64
#define EH      512            // expert hidden
#define SH      4096           // shared hidden
#define NLAYER  2
#define TOPK    4
#define TOTA    (NTOK*TOPK)
#define MAXTILES (NTOK/16)     // worst-case 16-row tiles per expert

typedef __bf16 bf16_t;
typedef __attribute__((ext_vector_type(16))) __bf16 v16bf;
typedef __attribute__((ext_vector_type(8)))  float  v8f;
typedef int v4i __attribute__((vector_size(16)));

#define AS1 __attribute__((address_space(1)))
#define AS3 __attribute__((address_space(3)))

#if defined(__AMDGCN__) && __has_builtin(__builtin_amdgcn_global_load_async_to_lds_b128)
#define HAVE_ASYNC 1
#else
#define HAVE_ASYNC 0
#endif

// ---------------- helpers ----------------
__device__ inline bf16_t f2bf(float f) {
    unsigned u = __builtin_bit_cast(unsigned, f);
    unsigned r = (u + 0x7FFFu + ((u >> 16) & 1u)) >> 16;   // RNE
    unsigned short h = (unsigned short)r;
    return __builtin_bit_cast(bf16_t, h);
}
__device__ inline float gelu_f(float x) {
    float x3 = x * x * x;
    return 0.5f * x * (1.0f + tanhf(0.7978845608028654f * (x + 0.044715f * x3)));
}
__device__ inline v8f wmma_bf16(v16bf a, v16bf b, v8f c) {
    return __builtin_amdgcn_wmma_f32_16x16x32_bf16(false, a, false, b, (short)0, c,
                                                   false, false);
}
// stage 16 bytes global -> LDS (async on CDNA5, sync fallback elsewhere)
__device__ inline void stage_a16(const bf16_t* g, bf16_t* l) {
#if HAVE_ASYNC
    __builtin_amdgcn_global_load_async_to_lds_b128((AS1 v4i*)g, (AS3 v4i*)l, 0, 0);
#else
    *(float4*)l = *(const float4*)g;
#endif
}
__device__ inline void async_fence() {
#if HAVE_ASYNC
#if __has_builtin(__builtin_amdgcn_s_wait_asynccnt)
    __builtin_amdgcn_s_wait_asynccnt(0);
#else
    asm volatile("s_wait_asynccnt 0" ::: "memory");
#endif
#endif
}

// ---------------- small utility kernels ----------------
__global__ void zero_u32(unsigned* p, int n) {
    int i = blockIdx.x * blockDim.x + threadIdx.x;
    if (i < n) p[i] = 0u;
}

__global__ void f32_to_bf16_k(const float* __restrict__ in, bf16_t* __restrict__ out, size_t n) {
    size_t i = (size_t)blockIdx.x * blockDim.x + threadIdx.x;
    if (i < n) out[i] = f2bf(in[i]);
}

// pack w[batch][K][Nc] (row-major) -> out[batch][Nc][K] bf16 (col-major of original)
__global__ void pack_colmajor(const float* __restrict__ w, bf16_t* __restrict__ out,
                              int K, int Nc, size_t total) {
    size_t i = (size_t)blockIdx.x * blockDim.x + threadIdx.x;
    if (i >= total) return;
    size_t mat  = (size_t)K * Nc;
    size_t bidx = i / mat;
    size_t rem  = i - bidx * mat;
    int n  = (int)(rem / K);
    int kk = (int)(rem - (size_t)n * K);
    out[bidx * mat + (size_t)n * K + kk] = f2bf(w[bidx * mat + (size_t)kk * Nc + n]);
}

// ---------------- layernorm (f32 in, f32 + optional bf16 out) ----------------
__global__ void ln_kernel(const float* __restrict__ x, const float* __restrict__ s,
                          const float* __restrict__ b, float* __restrict__ outf,
                          bf16_t* __restrict__ outb) {
    int n = blockIdx.x;
    int t = threadIdx.x;                  // 128 threads, 4 elems each
    __shared__ float red[128];
    const float* row = x + (size_t)n * DD;
    float v0 = row[t], v1 = row[t + 128], v2 = row[t + 256], v3 = row[t + 384];
    red[t] = v0 + v1 + v2 + v3;
    __syncthreads();
    for (int o = 64; o > 0; o >>= 1) { if (t < o) red[t] += red[t + o]; __syncthreads(); }
    float mean = red[0] * (1.0f / DD);
    __syncthreads();
    float d0 = v0 - mean, d1 = v1 - mean, d2 = v2 - mean, d3 = v3 - mean;
    red[t] = d0 * d0 + d1 * d1 + d2 * d2 + d3 * d3;
    __syncthreads();
    for (int o = 64; o > 0; o >>= 1) { if (t < o) red[t] += red[t + o]; __syncthreads(); }
    float rstd = rsqrtf(red[0] * (1.0f / DD) + 1e-5f);
    float y0 = d0 * rstd * s[t]       + b[t];
    float y1 = d1 * rstd * s[t + 128] + b[t + 128];
    float y2 = d2 * rstd * s[t + 256] + b[t + 256];
    float y3 = d3 * rstd * s[t + 384] + b[t + 384];
    float* orow = outf + (size_t)n * DD;
    orow[t] = y0; orow[t + 128] = y1; orow[t + 256] = y2; orow[t + 384] = y3;
    if (outb) {
        bf16_t* brow = outb + (size_t)n * DD;
        brow[t] = f2bf(y0); brow[t + 128] = f2bf(y1);
        brow[t + 256] = f2bf(y2); brow[t + 384] = f2bf(y3);
    }
}

// ---------------- WMMA GEMM: C[M,Ncols] = A[M,K] @ W + bias ----------------
// A: bf16 row-major [M,K]; Wp: bf16 packed col-major [Ncols,K].
// Block tile 64x256 (8 waves, 2x4), wave tile 32x64 (8 wmma/K-step).
// A staged via double-buffered async-to-LDS; B direct from global (L2-resident).
// EPI 0: store f32   EPI 1: gelu -> store bf16   EPI 2: atomicAdd f32 (residual)
template <int EPI>
__global__ __launch_bounds__(256)
void gemm_wmma(const bf16_t* __restrict__ A, const bf16_t* __restrict__ Wp,
               const float* __restrict__ bias, float* __restrict__ Cf,
               bf16_t* __restrict__ Cb, int M, int Ncols, int K) {
    constexpr int AST = 40;                       // LDS row stride (bf16): 80B = 5*16B
    __shared__ bf16_t atile[2][64 * AST];
    const int lane = threadIdx.x & 31;
    const int wv   = threadIdx.x >> 5;
    const int wm   = wv >> 2;                     // 0..1  (M sub-tile)
    const int wn   = wv & 3;                      // 0..3  (N sub-tile)
    const int tm   = blockIdx.x * 64;
    const int tn   = blockIdx.y * 256 + wn * 64;
    const int lr   = lane & 15;
    const int kh   = (lane >> 4) * 16;            // K-half select
    const bf16_t* bbase = Wp + (size_t)(tn + lr) * K + kh;
    // A stager: thread t copies 16B of row (t>>2), chunk (t&3)
    const int arow = threadIdx.x >> 2;            // 0..63
    const int acol = (threadIdx.x & 3) * 8;       // bf16 offset in 32-col chunk
    const bf16_t* ag = A + (size_t)(tm + arow) * K + acol;
    bf16_t* lbuf0 = &atile[0][arow * AST + acol];
    bf16_t* lbuf1 = &atile[1][arow * AST + acol];

    v8f acc[2][4];
    {
        v8f zz = {0, 0, 0, 0, 0, 0, 0, 0};
        for (int i = 0; i < 2; i++)
            for (int j = 0; j < 4; j++) acc[i][j] = zz;
    }

    stage_a16(ag, lbuf0);
    async_fence();
    __syncthreads();

    int cur = 0;
    for (int k0 = 0; k0 < K; k0 += 32) {
        if (k0 + 32 < K) stage_a16(ag + k0 + 32, cur ? lbuf0 : lbuf1);
        const bf16_t* at = atile[cur];
        v16bf a0 = *(const v16bf*)(at + (wm * 32 + lr) * AST + kh);
        v16bf a1 = *(const v16bf*)(at + (wm * 32 + 16 + lr) * AST + kh);
        v16bf b0 = *(const v16bf*)(bbase + k0);
        v16bf b1 = *(const v16bf*)(bbase + (size_t)16 * K + k0);
        v16bf b2 = *(const v16bf*)(bbase + (size_t)32 * K + k0);
        v16bf b3 = *(const v16bf*)(bbase + (size_t)48 * K + k0);
        acc[0][0] = wmma_bf16(a0, b0, acc[0][0]);
        acc[0][1] = wmma_bf16(a0, b1, acc[0][1]);
        acc[0][2] = wmma_bf16(a0, b2, acc[0][2]);
        acc[0][3] = wmma_bf16(a0, b3, acc[0][3]);
        acc[1][0] = wmma_bf16(a1, b0, acc[1][0]);
        acc[1][1] = wmma_bf16(a1, b1, acc[1][1]);
        acc[1][2] = wmma_bf16(a1, b2, acc[1][2]);
        acc[1][3] = wmma_bf16(a1, b3, acc[1][3]);
        async_fence();
        __syncthreads();
        cur ^= 1;
    }

    for (int mi = 0; mi < 2; mi++) {
        int rbase = tm + wm * 32 + mi * 16 + ((lane >> 4) << 3);
        for (int ni = 0; ni < 4; ni++) {
            int col  = tn + ni * 16 + lr;
            float bb = bias ? bias[col] : 0.0f;
            for (int r = 0; r < 8; r++) {
                int row = rbase + r;
                float v = acc[mi][ni][r] + bb;
                if (EPI == 0)      Cf[(size_t)row * Ncols + col] = v;
                else if (EPI == 1) Cb[(size_t)row * Ncols + col] = f2bf(gelu_f(v));
                else               atomicAdd(&Cf[(size_t)row * Ncols + col], v);
            }
        }
    }
}

// ---------------- flash attention (wave-per-row, dh=64 -> 2 f32/lane) --------
__global__ void att_kernel(const float* __restrict__ q, const float* __restrict__ k,
                           const float* __restrict__ v, float* __restrict__ o) {
    int bh = blockIdx.x;                  // BATCH*NHEAD
    int b  = bh >> 3, h = bh & 7;
    int wv = threadIdx.x >> 5, lane = threadIdx.x & 31;
    int s  = blockIdx.y * 8 + wv;
    size_t rq = ((size_t)(b * SEQ + s)) * DD + h * DHEAD + lane * 2;
    float2 qv = *(const float2*)(q + rq);
    float m = -1e30f, l = 0.0f, o0 = 0.0f, o1 = 0.0f;
    for (int kk = 0; kk < SEQ; kk++) {
        size_t rk = ((size_t)(b * SEQ + kk)) * DD + h * DHEAD + lane * 2;
        float2 kv = *(const float2*)(k + rk);
        float p = qv.x * kv.x + qv.y * kv.y;
        p += __shfl_xor(p, 16); p += __shfl_xor(p, 8); p += __shfl_xor(p, 4);
        p += __shfl_xor(p, 2);  p += __shfl_xor(p, 1);
        p *= 0.125f;                              // 1/sqrt(64)
        float mn = fmaxf(m, p);
        float c  = __expf(m - mn);
        float w  = __expf(p - mn);
        float2 vv = *(const float2*)(v + rk);
        l  = l * c + w;
        o0 = o0 * c + w * vv.x;
        o1 = o1 * c + w * vv.y;
        m  = mn;
    }
    float inv = 1.0f / l;
    o[rq] = o0 * inv; o[rq + 1] = o1 * inv;
}

// ---------------- MoE router: scores, softmax, top-4, aux stats --------------
__global__ void router_kernel(const float* __restrict__ lnf, const float* __restrict__ rd,
                              const float* __restrict__ ru, int* __restrict__ ti,
                              float* __restrict__ tw, int* __restrict__ counts,
                              float* __restrict__ accP, float* __restrict__ accZ) {
    int n = blockIdx.x, t = threadIdx.x;     // 64 threads
    __shared__ float sr[RRANK];
    __shared__ float ss[NEXP];
    __shared__ float sp[NEXP];
    const float* row = lnf + (size_t)n * DD;
    float a = 0.0f;
    for (int d = 0; d < DD; d++) a += row[d] * rd[(size_t)d * RRANK + t];
    sr[t] = a;
    __syncthreads();
    if (t < NEXP) {
        float sc = 0.0f;
        for (int r = 0; r < RRANK; r++) sc += sr[r] * ru[(size_t)r * NEXP + t];
        ss[t] = sc;
    }
    __syncthreads();
    if (t == 0) {
        float mx = ss[0];
        for (int e = 1; e < NEXP; e++) mx = fmaxf(mx, ss[e]);
        float se = 0.0f;
        for (int e = 0; e < NEXP; e++) { sp[e] = __expf(ss[e] - mx); se += sp[e]; }
        float inv = 1.0f / se;
        for (int e = 0; e < NEXP; e++) sp[e] *= inv;
        float lse = mx + __logf(se);
        atomicAdd(accZ, lse * lse);
    }
    __syncthreads();
    if (t < NEXP) atomicAdd(&accP[t], sp[t]);
    if (t == 0) {
        float pv[NEXP];
        for (int e = 0; e < NEXP; e++) pv[e] = sp[e];
        int sel[TOPK]; float sv[TOPK]; float ssum = 0.0f;
        for (int kk = 0; kk < TOPK; kk++) {
            int bi = 0; float bvv = -1.0f;
            for (int e = 0; e < NEXP; e++) if (pv[e] > bvv) { bvv = pv[e]; bi = e; }
            sel[kk] = bi; sv[kk] = bvv; ssum += bvv; pv[bi] = -2.0f;
        }
        float rinv = 1.0f / ssum;
        for (int kk = 0; kk < TOPK; kk++) {
            ti[n * TOPK + kk] = sel[kk];
            tw[n * TOPK + kk] = sv[kk] * rinv;
            atomicAdd(&counts[sel[kk]], 1);
        }
    }
}

__global__ void scan_kernel(const int* __restrict__ counts, int* __restrict__ offsets,
                            int* __restrict__ cursor) {
    if (threadIdx.x == 0) {
        int acc = 0;
        for (int e = 0; e < NEXP; e++) { offsets[e] = acc; acc += counts[e]; cursor[e] = 0; }
        offsets[NEXP] = acc;
    }
}

__global__ void scatter_kernel(const int* __restrict__ ti, const float* __restrict__ tw,
                               const int* __restrict__ offsets, int* __restrict__ cursor,
                               int* __restrict__ tok, float* __restrict__ gl) {
    int n = blockIdx.x * blockDim.x + threadIdx.x;
    if (n >= NTOK) return;
    for (int kk = 0; kk < TOPK; kk++) {
        int e   = ti[n * TOPK + kk];
        int pos = atomicAdd(&cursor[e], 1);
        int slot = offsets[e] + pos;
        tok[slot] = n;
        gl[slot]  = tw[n * TOPK + kk];
    }
}

// ---- expert FFN pass 1: h = gelu(gather(ln2) @ w1[e] + b1[e]) ----
// One block per 16-token tile; 8 waves x 64 cols cover all EH=512 columns.
__global__ __launch_bounds__(256)
void moe_ffn1(const bf16_t* __restrict__ lnb, const bf16_t* __restrict__ w1p,
              const float* __restrict__ b1, const int* __restrict__ counts,
              const int* __restrict__ offsets, const int* __restrict__ tok,
              bf16_t* __restrict__ hcomp) {
    int e    = blockIdx.x / MAXTILES;
    int tIdx = blockIdx.x % MAXTILES;
    int cnt  = counts[e];
    if (tIdx * 16 >= cnt) return;
    int lane = threadIdx.x & 31, wv = threadIdx.x >> 5;
    int tn = wv * 64;
    int lr = lane & 15;
    int kh = (lane >> 4) * 16;
    int cl = min(tIdx * 16 + lr, cnt - 1);
    int token = tok[offsets[e] + cl];
    const bf16_t* ap = lnb + (size_t)token * DD + kh;
    const bf16_t* bp = w1p + (size_t)e * DD * EH + (size_t)(tn + lr) * DD + kh;
    v8f acc[4];
    {
        v8f zz = {0, 0, 0, 0, 0, 0, 0, 0};
        for (int j = 0; j < 4; j++) acc[j] = zz;
    }
    for (int k0 = 0; k0 < DD; k0 += 32) {
        v16bf a = *(const v16bf*)(ap + k0);
        v16bf b0 = *(const v16bf*)(bp + k0);
        v16bf b1f = *(const v16bf*)(bp + (size_t)16 * DD + k0);
        v16bf b2 = *(const v16bf*)(bp + (size_t)32 * DD + k0);
        v16bf b3 = *(const v16bf*)(bp + (size_t)48 * DD + k0);
        acc[0] = wmma_bf16(a, b0, acc[0]);
        acc[1] = wmma_bf16(a, b1f, acc[1]);
        acc[2] = wmma_bf16(a, b2, acc[2]);
        acc[3] = wmma_bf16(a, b3, acc[3]);
    }
    int rb = (lane >> 4) << 3;
    for (int j = 0; j < 4; j++) {
        int col  = tn + j * 16 + lr;
        float bb = b1[e * EH + col];
        for (int r = 0; r < 8; r++) {
            int rowt = tIdx * 16 + rb + r;
            if (rowt < cnt) {
                int slot = offsets[e] + rowt;
                hcomp[(size_t)slot * EH + col] = f2bf(gelu_f(acc[j][r] + bb));
            }
        }
    }
}

// ---- expert FFN pass 2: x += gate * (h @ w2[e] + b2[e]) ----
__global__ __launch_bounds__(256)
void moe_ffn2(const bf16_t* __restrict__ hcomp, const bf16_t* __restrict__ w2p,
              const float* __restrict__ b2, const int* __restrict__ counts,
              const int* __restrict__ offsets, const int* __restrict__ tok,
              const float* __restrict__ gl, float* __restrict__ xcur) {
    int e    = blockIdx.x / MAXTILES;
    int tIdx = blockIdx.x % MAXTILES;
    int cnt  = counts[e];
    if (tIdx * 16 >= cnt) return;
    int lane = threadIdx.x & 31, wv = threadIdx.x >> 5;
    int tn = wv * 64;
    int lr = lane & 15;
    int kh = (lane >> 4) * 16;
    int cl = min(tIdx * 16 + lr, cnt - 1);
    const bf16_t* ap = hcomp + (size_t)(offsets[e] + cl) * EH + kh;
    const bf16_t* bp = w2p + (size_t)e * EH * DD + (size_t)(tn + lr) * EH + kh;
    v8f acc[4];
    {
        v8f zz = {0, 0, 0, 0, 0, 0, 0, 0};
        for (int j = 0; j < 4; j++) acc[j] = zz;
    }
    for (int k0 = 0; k0 < EH; k0 += 32) {
        v16bf a = *(const v16bf*)(ap + k0);
        v16bf b0 = *(const v16bf*)(bp + k0);
        v16bf b1f = *(const v16bf*)(bp + (size_t)16 * EH + k0);
        v16bf b2 = *(const v16bf*)(bp + (size_t)32 * EH + k0);
        v16bf b3 = *(const v16bf*)(bp + (size_t)48 * EH + k0);
        acc[0] = wmma_bf16(a, b0, acc[0]);
        acc[1] = wmma_bf16(a, b1f, acc[1]);
        acc[2] = wmma_bf16(a, b2, acc[2]);
        acc[3] = wmma_bf16(a, b3, acc[3]);
    }
    int rb = (lane >> 4) << 3;
    for (int j = 0; j < 4; j++) {
        int col  = tn + j * 16 + lr;
        float bb = b2[e * DD + col];
        for (int r = 0; r < 8; r++) {
            int rowt = tIdx * 16 + rb + r;
            if (rowt < cnt) {
                int slot  = offsets[e] + rowt;
                int token = tok[slot];
                float g   = gl[slot];
                atomicAdd(&xcur[(size_t)token * DD + col], (acc[j][r] + bb) * g);
            }
        }
    }
}

// ---------------- aux losses -------------------------------------------------
__global__ void aux_finalize(const int* __restrict__ counts, const float* __restrict__ accP,
                             const float* __restrict__ accZ, float* __restrict__ out_lb,
                             float* __restrict__ out_z) {
    float lb = 0.0f;
    const float invN = 1.0f / (float)NTOK;
    for (int e = 0; e < NEXP; e++)
        lb += ((float)counts[e] * invN) * (accP[e] * invN);
    *out_lb = lb * ((float)NEXP / (float)TOPK);
    *out_z  = accZ[0] * invN;
}

// ---------------- host launch ----------------
extern "C" void kernel_launch(void* const* d_in, const int* in_sizes, int n_in,
                              void* d_out, int out_size, void* d_ws, size_t ws_size,
                              hipStream_t stream) {
    (void)in_sizes; (void)n_in; (void)out_size; (void)ws_size;
    const float* x     = (const float*)d_in[0];
    const float* ln1_s = (const float*)d_in[1];
    const float* ln1_b = (const float*)d_in[2];
    const float* ln2_s = (const float*)d_in[3];
    const float* ln2_b = (const float*)d_in[4];
    const float* wq    = (const float*)d_in[5];
    const float* bq    = (const float*)d_in[6];
    const float* wk    = (const float*)d_in[7];
    const float* bk    = (const float*)d_in[8];
    const float* wvv   = (const float*)d_in[9];
    const float* bv    = (const float*)d_in[10];
    const float* wo    = (const float*)d_in[11];
    const float* bo    = (const float*)d_in[12];
    const float* rd    = (const float*)d_in[13];
    const float* ru    = (const float*)d_in[14];
    const float* w1    = (const float*)d_in[15];
    const float* b1    = (const float*)d_in[16];
    const float* w2    = (const float*)d_in[17];
    const float* b2    = (const float*)d_in[18];
    const float* ws1   = (const float*)d_in[19];
    const float* bs1   = (const float*)d_in[20];
    const float* ws2   = (const float*)d_in[21];
    const float* bs2   = (const float*)d_in[22];
    const float* fn_s  = (const float*)d_in[23];
    const float* fn_b  = (const float*)d_in[24];

    const size_t ND = (size_t)NTOK * DD;
    size_t off = 0;
    auto alloc = [&](size_t bytes) -> void* {
        off = (off + 255) & ~(size_t)255;
        void* p = (char*)d_ws + off;
        off += bytes;
        return p;
    };
    float*  xcur  = (float*)alloc(ND * 4);
    float*  lnf   = (float*)alloc(ND * 4);
    bf16_t* lnb   = (bf16_t*)alloc(ND * 2);
    float*  qf    = (float*)alloc(ND * 4);
    float*  kf    = (float*)alloc(ND * 4);
    float*  vf    = (float*)alloc(ND * 4);
    float*  aof   = (float*)alloc(ND * 4);
    bf16_t* aob   = (bf16_t*)alloc(ND * 2);
    bf16_t* hsb   = (bf16_t*)alloc((size_t)NTOK * SH * 2);
    bf16_t* hcomp = (bf16_t*)alloc((size_t)TOTA * EH * 2);
    bf16_t* wpQ   = (bf16_t*)alloc((size_t)DD * DD * 2);
    bf16_t* wpK   = (bf16_t*)alloc((size_t)DD * DD * 2);
    bf16_t* wpV   = (bf16_t*)alloc((size_t)DD * DD * 2);
    bf16_t* wpO   = (bf16_t*)alloc((size_t)DD * DD * 2);
    bf16_t* wpS1  = (bf16_t*)alloc((size_t)DD * SH * 2);
    bf16_t* wpS2  = (bf16_t*)alloc((size_t)SH * DD * 2);
    bf16_t* wpE1  = (bf16_t*)alloc((size_t)NEXP * DD * EH * 2);
    bf16_t* wpE2  = (bf16_t*)alloc((size_t)NEXP * EH * DD * 2);
    int*    ti    = (int*)alloc((size_t)NTOK * TOPK * 4);
    float*  tw    = (float*)alloc((size_t)NTOK * TOPK * 4);
    int*    counts = (int*)alloc((size_t)(2 * NEXP + 1) * 4);
    float*  accP   = (float*)(counts + NEXP);
    float*  accZ   = accP + NEXP;
    int*    offsets = (int*)alloc((size_t)(NEXP + 1) * 4);
    int*    cursor  = (int*)alloc((size_t)NEXP * 4);
    int*    tok     = (int*)alloc((size_t)TOTA * 4);
    float*  gl      = (float*)alloc((size_t)TOTA * 4);

    float* outF = (float*)d_out;

    (void)hipMemcpyAsync(xcur, x, ND * 4, hipMemcpyDeviceToDevice, stream);

    const dim3 gD(NTOK / 64, DD / 256);    // GEMM grid, Ncols = D
    auto packGrid = [](size_t total) { return (unsigned)((total + 255) / 256); };

    for (int l = 0; l < NLAYER; l++) {
        // ---- pack this layer's weights to bf16 col-major ----
        size_t tDD = (size_t)DD * DD;
        pack_colmajor<<<packGrid(tDD), 256, 0, stream>>>(wq + l * tDD, wpQ, DD, DD, tDD);
        pack_colmajor<<<packGrid(tDD), 256, 0, stream>>>(wk + l * tDD, wpK, DD, DD, tDD);
        pack_colmajor<<<packGrid(tDD), 256, 0, stream>>>(wvv + l * tDD, wpV, DD, DD, tDD);
        pack_colmajor<<<packGrid(tDD), 256, 0, stream>>>(wo + l * tDD, wpO, DD, DD, tDD);
        size_t tS = (size_t)DD * SH;
        pack_colmajor<<<packGrid(tS), 256, 0, stream>>>(ws1 + l * tS, wpS1, DD, SH, tS);
        pack_colmajor<<<packGrid(tS), 256, 0, stream>>>(ws2 + l * tS, wpS2, SH, DD, tS);
        size_t tE = (size_t)NEXP * DD * EH;
        pack_colmajor<<<packGrid(tE), 256, 0, stream>>>(w1 + l * tE, wpE1, DD, EH, tE);
        pack_colmajor<<<packGrid(tE), 256, 0, stream>>>(w2 + l * tE, wpE2, EH, DD, tE);

        // ---- attention ----
        ln_kernel<<<NTOK, 128, 0, stream>>>(xcur, ln1_s + l * DD, ln1_b + l * DD, lnf, lnb);
        gemm_wmma<0><<<gD, 256, 0, stream>>>(lnb, wpQ, bq + l * DD, qf, nullptr, NTOK, DD, DD);
        gemm_wmma<0><<<gD, 256, 0, stream>>>(lnb, wpK, bk + l * DD, kf, nullptr, NTOK, DD, DD);
        gemm_wmma<0><<<gD, 256, 0, stream>>>(lnb, wpV, bv + l * DD, vf, nullptr, NTOK, DD, DD);
        att_kernel<<<dim3(BATCH * NHEAD, SEQ / 8), 256, 0, stream>>>(qf, kf, vf, aof);
        f32_to_bf16_k<<<packGrid(ND), 256, 0, stream>>>(aof, aob, ND);
        gemm_wmma<2><<<gD, 256, 0, stream>>>(aob, wpO, bo + l * DD, xcur, nullptr, NTOK, DD, DD);

        // ---- MoE ----
        ln_kernel<<<NTOK, 128, 0, stream>>>(xcur, ln2_s + l * DD, ln2_b + l * DD, lnf, lnb);
        zero_u32<<<1, 128, 0, stream>>>((unsigned*)counts, 2 * NEXP + 1);
        router_kernel<<<NTOK, 64, 0, stream>>>(lnf, rd + (size_t)l * DD * RRANK,
                                               ru + (size_t)l * RRANK * NEXP,
                                               ti, tw, counts, accP, accZ);
        scan_kernel<<<1, 32, 0, stream>>>(counts, offsets, cursor);
        scatter_kernel<<<NTOK / 256, 256, 0, stream>>>(ti, tw, offsets, cursor, tok, gl);
        moe_ffn1<<<NEXP * MAXTILES, 256, 0, stream>>>(
            lnb, wpE1, b1 + (size_t)l * NEXP * EH, counts, offsets, tok, hcomp);
        moe_ffn2<<<NEXP * MAXTILES, 256, 0, stream>>>(
            hcomp, wpE2, b2 + (size_t)l * NEXP * DD, counts, offsets, tok, gl, xcur);

        // ---- shared FFN ----
        gemm_wmma<1><<<dim3(NTOK / 64, SH / 256), 256, 0, stream>>>(
            lnb, wpS1, bs1 + (size_t)l * SH, nullptr, hsb, NTOK, SH, DD);
        gemm_wmma<2><<<gD, 256, 0, stream>>>(hsb, wpS2, bs2 + (size_t)l * DD, xcur, nullptr,
                                             NTOK, DD, SH);

        // ---- aux losses (lbs at ND+l, zs at ND+2+l) ----
        aux_finalize<<<1, 1, 0, stream>>>(counts, accP, accZ, outF + ND + l, outF + ND + 2 + l);
    }

    // ---- final layernorm straight into d_out ----
    ln_kernel<<<NTOK, 128, 0, stream>>>(xcur, fn_s, fn_b, outF, nullptr);
}
